// TFunc_21947282883078
// MI455X (gfx1250) — matrix operand order repaired and checked by reading, not compiled, not run on hardware
//
#include <hip/hip_runtime.h>
#include <hip/hip_bf16.h>
#include <math.h>

// ---------------------------------------------------------------------------
// Problem constants (from reference)
// ---------------------------------------------------------------------------
#define KNBR 32
#define IN_LEN 16
#define OUT_LEN 25
#define EMB 32
#define ENC 64
#define DYN 32
#define DEC 128
#define NEG_SLOPE 0.1f

typedef _Float16 v16h __attribute__((ext_vector_type(16)));
typedef float v8f __attribute__((ext_vector_type(8)));

__device__ __forceinline__ float lrelu(float x) { return x > 0.f ? x : NEG_SLOPE * x; }
__device__ __forceinline__ float sigm(float x) { return 1.f / (1.f + __expf(-x)); }

// D = A(16x32 f16) * B(32x16 f16) + C (16x16 f32)
__device__ __forceinline__ v8f wmma_f16(v16h a, v16h b, v8f c) {
  return __builtin_amdgcn_wmma_f32_16x16x32_f16(false, a, false, b, (short)0, c,
                                                false, false);
}

// A fragment from row-major f16 LDS tile: rows = 16, K chunk of 32 starting at
// `base` (base already offset to row0,k0), row stride `stride` elements.
// ISA layout: lane r<16: K {0..7,16..23}; lane r+16: K {8..15,24..31}.
__device__ __forceinline__ v16h load_A_lds(const _Float16* base, int stride, int lane) {
  int r = lane & 15, half = lane >> 4;
  v16h a;
#pragma unroll
  for (int j = 0; j < 8; ++j) {
    int k = ((j & 4) ? 16 : 0) + 8 * half + (j & 3) * 2;
    a[2 * j] = base[r * stride + k];
    a[2 * j + 1] = base[r * stride + k + 1];
  }
  return a;
}

// B fragment from global row-major f32 weight W[N][Kdim] (out-major, like
// torch Linear weight). Columns of B = rows of W. Tile cols [colbase,colbase+16),
// K chunk [kb, kb+32). Lane half h holds K {8h..8h+7, 16+8h..16+8h+7} within the chunk.
__device__ __forceinline__ v16h load_B_gw(const float* W, int Kdim, int colbase,
                                          int kb, int lane) {
  int n = colbase + (lane & 15), half = lane >> 4;
  v16h b;
#pragma unroll
  for (int j = 0; j < 8; ++j) {
    int k = kb + ((j & 4) ? 16 : 0) + 8 * half + (j & 3) * 2;
    b[2 * j] = (_Float16)W[n * Kdim + k];
    b[2 * j + 1] = (_Float16)W[n * Kdim + k + 1];
  }
  return b;
}

// ---------------------------------------------------------------------------
// Kernel 1: neighbor embedding + per-hero max pool + dyn projection.
// One wave per hero (4 heroes / 128-thread block).
// enc layout: [B][96] = (hist_dyn | scene_dyn | fut_dyn); writes cols 32..63.
// ---------------------------------------------------------------------------
__global__ void __launch_bounds__(128)
scene_kernel(const float* __restrict__ hist, const float* __restrict__ nbrs,
             const float* __restrict__ nbrW, const float* __restrict__ nbrb,
             const float* __restrict__ dynW, const float* __restrict__ dynb,
             float* __restrict__ enc) {
  __shared__ _Float16 rel[4][KNBR][32];  // relative tiles (f16), per hero
  __shared__ float sp[4][ENC];           // pooled scene features

  int tid = threadIdx.x;
  int wave = tid >> 5, lane = tid & 31;
  int hero = blockIdx.x * 4 + wave;

  // Stage relative = hist[hero] - nbrs[hero*32 + i]  (lane = column)
  float hv = hist[hero * 32 + lane];
#pragma unroll 4
  for (int i = 0; i < KNBR; ++i)
    rel[wave][i][lane] = (_Float16)(hv - nbrs[(hero * KNBR + i) * 32 + lane]);

  v16h a0 = load_A_lds(&rel[wave][0][0], 32, lane);   // nbr rows 0..15
  v16h a1 = load_A_lds(&rel[wave][16][0], 32, lane);  // nbr rows 16..31

#pragma unroll
  for (int t = 0; t < 4; ++t) {  // 4 column tiles of 16 -> 64 outputs
    v16h b = load_B_gw(nbrW, 32, 16 * t, 0, lane);
    v8f c0 = {}, c1 = {};
    c0 = wmma_f16(a0, b, c0);
    c1 = wmma_f16(a1, b, c1);
    float m = -3.0e38f;
#pragma unroll
    for (int v = 0; v < 8; ++v) m = fmaxf(m, fmaxf(c0[v], c1[v]));
    m = fmaxf(m, __shfl_xor(m, 16, 32));  // fold the two M-halves
    if (lane < 16) {
      int col = 16 * t + lane;
      sp[wave][col] = lrelu(m + nbrb[col]);  // bias+lrelu commute with max
    }
  }
  __syncthreads();

  // scene_dyn = lrelu(sp @ dynW.T + dynb): 4 heroes x 32 outs = 128 threads
  int h = tid >> 5, d = tid & 31;
  float s = dynb[d];
#pragma unroll 8
  for (int k = 0; k < ENC; ++k) s += sp[h][k] * dynW[d * ENC + k];
  enc[(blockIdx.x * 4 + h) * 96 + 32 + d] = lrelu(s);
}

// ---------------------------------------------------------------------------
// Kernel 2: encoder LSTM (EMB=32 -> ENC=64) over T steps + dyn projection.
// 16 heroes per 128-thread block (4 waves). Gate columns: 4*ENC = 256 = 16
// tiles; wave w owns tiles {w, w+4, w+8, w+12} = cols [16w,16w+16) of each of
// i/f/g/o -> gate combine + cell state stay in registers.
// ---------------------------------------------------------------------------
__global__ void __launch_bounds__(128)
encoder_kernel(const float* __restrict__ x, int T, int outcol,
               const float* __restrict__ ipW, const float* __restrict__ ipb,
               const float* __restrict__ Wih, const float* __restrict__ Whh,
               const float* __restrict__ bih, const float* __restrict__ bhh,
               const float* __restrict__ dynW, const float* __restrict__ dynb,
               float* __restrict__ enc) {
  __shared__ _Float16 emb[16][EMB];
  __shared__ _Float16 hh[16][ENC];
  __shared__ float hf[16][ENC];
  __shared__ float bs[4 * ENC];

  int tid = threadIdx.x;
  int wave = tid >> 5, lane = tid & 31;
  int heroBase = blockIdx.x * 16;
  int xs = 2 * T;  // per-hero row stride in x

  for (int i = tid; i < 4 * ENC; i += 128) bs[i] = bih[i] + bhh[i];
  for (int i = tid; i < 16 * ENC; i += 128) (&hh[0][0])[i] = (_Float16)0.f;

  // Preload weight B-fragments (constant over time)
  v16h Bih[4], Bhh[4][2];
#pragma unroll
  for (int q = 0; q < 4; ++q) {
    int tile = wave + 4 * q;
    Bih[q] = load_B_gw(Wih, EMB, 16 * tile, 0, lane);
    Bhh[q][0] = load_B_gw(Whh, ENC, 16 * tile, 0, lane);
    Bhh[q][1] = load_B_gw(Whh, ENC, 16 * tile, 32, lane);
  }
  v8f cst = {};  // cell state for (16 heroes) x (units 16w..16w+15)
  __syncthreads();

  for (int t = 0; t < T; ++t) {
    // Input embedding: emb = lrelu(x_t @ ipW.T + ipb)
    for (int i = tid; i < 16 * EMB; i += 128) {
      int m = i >> 5, e = i & 31;
      const float* xr = x + (heroBase + m) * xs + 2 * t;
      emb[m][e] = (_Float16)lrelu(xr[0] * ipW[2 * e] + xr[1] * ipW[2 * e + 1] + ipb[e]);
    }
    __syncthreads();
    v16h Ae = load_A_lds(&emb[0][0], EMB, lane);
    v16h Ah0 = load_A_lds(&hh[0][0], ENC, lane);
    v16h Ah1 = load_A_lds(&hh[0][32], ENC, lane);
    __syncthreads();  // all A-fragment reads done before hh is rewritten

    v8f C[4];
#pragma unroll
    for (int q = 0; q < 4; ++q) {
      float b = bs[16 * (wave + 4 * q) + (lane & 15)];
      v8f c;
#pragma unroll
      for (int v = 0; v < 8; ++v) c[v] = b;
      c = wmma_f16(Ae, Bih[q], c);
      c = wmma_f16(Ah0, Bhh[q][0], c);
      c = wmma_f16(Ah1, Bhh[q][1], c);
      C[q] = c;
    }
    // Gate combine fully in registers (identical lane/vgpr mapping per gate)
    int n = 16 * wave + (lane & 15);
#pragma unroll
    for (int v = 0; v < 8; ++v) {
      float ig = sigm(C[0][v]), fg = sigm(C[1][v]);
      float gg = tanhf(C[2][v]), og = sigm(C[3][v]);
      float cn = fg * cst[v] + ig * gg;
      cst[v] = cn;
      float hvv = og * tanhf(cn);
      int m = v + 8 * (lane >> 4);
      hh[m][n] = (_Float16)hvv;
      if (t == T - 1) hf[m][n] = hvv;
    }
    __syncthreads();
  }

  // dyn projection of final h: lrelu(h @ dynW.T + dynb) -> enc[:, outcol..+32)
  for (int i = tid; i < 16 * DYN; i += 128) {
    int m = i >> 5, d = i & 31;
    float s = dynb[d];
#pragma unroll 8
    for (int k = 0; k < ENC; ++k) s += hf[m][k] * dynW[d * ENC + k];
    enc[(heroBase + m) * 96 + outcol + d] = lrelu(s);
  }
}

// ---------------------------------------------------------------------------
// Kernel 3: decoder LSTM (3*DYN=96 -> DEC=128), 25 steps, constant input.
// 16 heroes per 256-thread block (8 waves). Gate cols: 4*DEC = 512 = 32 tiles;
// wave w owns tiles {w, w+8, w+16, w+24}. xg precomputed once into LDS.
// Output projection: 16 threads per hero (8 MACs each) + shfl-xor reduction.
// ---------------------------------------------------------------------------
__global__ void __launch_bounds__(256)
decoder_kernel(const float* __restrict__ enc, const float* __restrict__ Wih,
               const float* __restrict__ Whh, const float* __restrict__ bih,
               const float* __restrict__ bhh, const float* __restrict__ opW,
               const float* __restrict__ opb, float* __restrict__ out) {
  __shared__ _Float16 encT[16][96];
  __shared__ float xg[16][4 * DEC];  // 32 KB: input gates + biases (constant)
  __shared__ _Float16 hh[16][DEC];
  __shared__ float hf[16][DEC];
  __shared__ float bs[4 * DEC];

  int tid = threadIdx.x;
  int wave = tid >> 5, lane = tid & 31;
  int heroBase = blockIdx.x * 16;

  for (int i = tid; i < 4 * DEC; i += 256) bs[i] = bih[i] + bhh[i];
  for (int i = tid; i < 16 * 96; i += 256)
    (&encT[0][0])[i] = (_Float16)enc[heroBase * 96 + i];
  for (int i = tid; i < 16 * DEC; i += 256) (&hh[0][0])[i] = (_Float16)0.f;
  __syncthreads();

  // xg = enc @ Wih.T + (bih + bhh)  (K = 96 -> 3 chunks)
  v16h Ae[3];
#pragma unroll
  for (int ck = 0; ck < 3; ++ck) Ae[ck] = load_A_lds(&encT[0][32 * ck], 96, lane);
#pragma unroll
  for (int q = 0; q < 4; ++q) {
    int tile = wave + 8 * q;
    float b = bs[16 * tile + (lane & 15)];
    v8f c;
#pragma unroll
    for (int v = 0; v < 8; ++v) c[v] = b;
#pragma unroll
    for (int ck = 0; ck < 3; ++ck) {
      v16h bw = load_B_gw(Wih, 96, 16 * tile, 32 * ck, lane);
      c = wmma_f16(Ae[ck], bw, c);
    }
#pragma unroll
    for (int v = 0; v < 8; ++v)
      xg[v + 8 * (lane >> 4)][16 * tile + (lane & 15)] = c[v];
  }

  // Preload recurrent weight fragments: 4 gate tiles x 4 K-chunks per wave
  v16h Bh[4][4];
#pragma unroll
  for (int q = 0; q < 4; ++q)
#pragma unroll
    for (int ck = 0; ck < 4; ++ck)
      Bh[q][ck] = load_B_gw(Whh, DEC, 16 * (wave + 8 * q), 32 * ck, lane);

  // Output-projection weights: thread tid covers hero hm = tid>>4, k-range
  // [8*(tid&15), 8*(tid&15)+8) for both output components. Hoisted to regs.
  int hm = tid >> 4, sub = tid & 15;
  float w0[8], w1[8];
#pragma unroll
  for (int kk = 0; kk < 8; ++kk) {
    w0[kk] = opW[sub * 8 + kk];
    w1[kk] = opW[DEC + sub * 8 + kk];
  }
  float ob0 = opb[0], ob1 = opb[1];

  v8f cst = {};
  __syncthreads();

  for (int t = 0; t < OUT_LEN; ++t) {
    v16h Ah[4];
#pragma unroll
    for (int ck = 0; ck < 4; ++ck) Ah[ck] = load_A_lds(&hh[0][32 * ck], DEC, lane);
    __syncthreads();

    v8f C[4];
#pragma unroll
    for (int q = 0; q < 4; ++q) {
      int tile = wave + 8 * q;
      v8f c;
#pragma unroll
      for (int v = 0; v < 8; ++v)
        c[v] = xg[v + 8 * (lane >> 4)][16 * tile + (lane & 15)];
#pragma unroll
      for (int ck = 0; ck < 4; ++ck) c = wmma_f16(Ah[ck], Bh[q][ck], c);
      C[q] = c;
    }
    int n = 16 * wave + (lane & 15);
#pragma unroll
    for (int v = 0; v < 8; ++v) {
      float ig = sigm(C[0][v]), fg = sigm(C[1][v]);
      float gg = tanhf(C[2][v]), og = sigm(C[3][v]);
      float cn = fg * cst[v] + ig * gg;
      cst[v] = cn;
      float hvv = og * tanhf(cn);
      int m = v + 8 * (lane >> 4);
      hh[m][n] = (_Float16)hvv;
      hf[m][n] = hvv;
    }
    __syncthreads();

    // fut_pred[t] = h @ opW.T + opb: 16 threads/hero, 8 MACs each per output,
    // then 4 shfl-xor rounds (masks 8/4/2/1 stay within the 16-thread group).
    float s0 = 0.f, s1 = 0.f;
#pragma unroll
    for (int kk = 0; kk < 8; ++kk) {
      float hvv = hf[hm][sub * 8 + kk];
      s0 += hvv * w0[kk];
      s1 += hvv * w1[kk];
    }
#pragma unroll
    for (int off = 8; off > 0; off >>= 1) {
      s0 += __shfl_xor(s0, off, 32);
      s1 += __shfl_xor(s1, off, 32);
    }
    if (sub == 0) {
      float* o = out + (heroBase + hm) * (2 * OUT_LEN) + 2 * t;
      o[0] = s0 + ob0;
      o[1] = s1 + ob1;
    }
    __syncthreads();
  }
}

// ---------------------------------------------------------------------------
extern "C" void kernel_launch(void* const* d_in, const int* in_sizes, int n_in,
                              void* d_out, int out_size, void* d_ws, size_t ws_size,
                              hipStream_t stream) {
  const float* hist = (const float*)d_in[0];
  const float* nbrs = (const float*)d_in[1];
  const float* fut = (const float*)d_in[2];
  // d_in[3] = segment_ids: repeat(arange(B), K) -> implicit contiguous layout
  const float* ipW = (const float*)d_in[4];
  const float* ipb = (const float*)d_in[5];
  const float* encWih = (const float*)d_in[6];
  const float* encWhh = (const float*)d_in[7];
  const float* encbih = (const float*)d_in[8];
  const float* encbhh = (const float*)d_in[9];
  const float* nbrW = (const float*)d_in[10];
  const float* nbrb = (const float*)d_in[11];
  const float* dynW = (const float*)d_in[12];
  const float* dynb = (const float*)d_in[13];
  const float* decWih = (const float*)d_in[14];
  const float* decWhh = (const float*)d_in[15];
  const float* decbih = (const float*)d_in[16];
  const float* decbhh = (const float*)d_in[17];
  const float* opW = (const float*)d_in[18];
  const float* opb = (const float*)d_in[19];

  int B = in_sizes[0] / (2 * IN_LEN);  // 20000
  float* enc = (float*)d_ws;           // [B][96] = hist_dyn|scene_dyn|fut_dyn
  float* outp = (float*)d_out;

  scene_kernel<<<B / 4, 128, 0, stream>>>(hist, nbrs, nbrW, nbrb, dynW, dynb, enc);
  encoder_kernel<<<B / 16, 128, 0, stream>>>(hist, IN_LEN, 0, ipW, ipb, encWih,
                                             encWhh, encbih, encbhh, dynW, dynb, enc);
  encoder_kernel<<<B / 16, 128, 0, stream>>>(fut, OUT_LEN, 64, ipW, ipb, encWih,
                                             encWhh, encbih, encbhh, dynW, dynb, enc);
  decoder_kernel<<<B / 16, 256, 0, stream>>>(enc, decWih, decWhh, decbih, decbhh,
                                             opW, opb, outp);
}
